// LorentzMLR_68341519614822
// MI455X (gfx1250) — matrix operand order repaired and checked by reading, not compiled, not run on hardware
//
#include <hip/hip_runtime.h>
#include <hip/hip_bf16.h>

// ---------------------------------------------------------------------------
// Lorentz MLR head:  logits = -arccosh(clip(x0*c0 - xf @ cs.T, 1+eps))
//   M = B*T = 4096, N = C = 32000, K = D = 256
// bf16 WMMA (f32 accumulate) GEMM, async-to-LDS double-buffered staging,
// fused hyperbolic epilogue with nontemporal stores.
// ---------------------------------------------------------------------------

typedef __attribute__((ext_vector_type(16))) __bf16 bf16x16;
typedef __attribute__((ext_vector_type(8)))  __bf16 bf16x8;
typedef __attribute__((ext_vector_type(8)))  float  f32x8;
typedef __attribute__((ext_vector_type(4)))  int    v4i;

#define AS1 __attribute__((address_space(1)))
#define AS3 __attribute__((address_space(3)))

#define M_DIM 4096
#define N_DIM 32000
#define K_DIM 256
#define W_STRIDE 257
#define EPS_1 1.000001f

#define LDS_STRIDE 40                 // bf16 elems per LDS tile row (32 + 8 pad)
#define TILE_ELEMS (128 * LDS_STRIDE) // one 128x32 tile (padded) = 10240 B

// Async DMA: 16 bytes per lane, global -> LDS, tracked by ASYNCcnt.
__device__ __forceinline__ void async_copy_b128(const __bf16* g, __bf16* l) {
#if __has_builtin(__builtin_amdgcn_global_load_async_to_lds_b128)
    __builtin_amdgcn_global_load_async_to_lds_b128(
        (AS1 v4i*)g, (AS3 v4i*)l, /*offset=*/0, /*cpol=*/0);
#else
    unsigned lds_off = (unsigned)(unsigned long long)(AS3 void*)l;
    asm volatile("global_load_async_to_lds_b128 %0, %1, off"
                 :: "v"(lds_off), "v"(g) : "memory");
#endif
}

__device__ __forceinline__ void wait_async_le4() {
#if __has_builtin(__builtin_amdgcn_s_wait_asynccnt)
    __builtin_amdgcn_s_wait_asynccnt(4);
#else
    asm volatile("s_wait_asynccnt 0x4" ::: "memory");
#endif
}
__device__ __forceinline__ void wait_async_0() {
#if __has_builtin(__builtin_amdgcn_s_wait_asynccnt)
    __builtin_amdgcn_s_wait_asynccnt(0);
#else
    asm volatile("s_wait_asynccnt 0x0" ::: "memory");
#endif
}

// ---------------------------------------------------------------------------
// prep_x: f32 x (4096x256) -> bf16 A (row-major), x0[m] = sqrt(1+||row||^2)
// ---------------------------------------------------------------------------
__global__ __launch_bounds__(256) void lorentz_prep_x(
    const float* __restrict__ x, __bf16* __restrict__ Abf, float* __restrict__ x0)
{
    const int wid  = threadIdx.x >> 5;
    const int lane = threadIdx.x & 31;
    const int row  = blockIdx.x * 8 + wid;          // 512 blocks * 8 = 4096

    const float* src = x + (size_t)row * K_DIM + lane * 8;
    float4 v0 = ((const float4*)src)[0];
    float4 v1 = ((const float4*)src)[1];

    float s = v0.x*v0.x + v0.y*v0.y + v0.z*v0.z + v0.w*v0.w
            + v1.x*v1.x + v1.y*v1.y + v1.z*v1.z + v1.w*v1.w;

    bf16x8 o;
    o[0] = (__bf16)v0.x; o[1] = (__bf16)v0.y; o[2] = (__bf16)v0.z; o[3] = (__bf16)v0.w;
    o[4] = (__bf16)v1.x; o[5] = (__bf16)v1.y; o[6] = (__bf16)v1.z; o[7] = (__bf16)v1.w;
    *(bf16x8*)(Abf + (size_t)row * K_DIM + lane * 8) = o;

    #pragma unroll
    for (int off = 16; off > 0; off >>= 1)
        s += __shfl_xor(s, off, 32);
    if (lane == 0)
        x0[row] = __fsqrt_rn(1.0f + s);
}

// ---------------------------------------------------------------------------
// prep_w: f32 lt_weight[:,1:] (32000 rows, stride 257, 4B-aligned source ->
// scalar loads) -> bf16 B (row-major 32000x256), c0[n] = sqrt(1+||row||^2)
// ---------------------------------------------------------------------------
__global__ __launch_bounds__(256) void lorentz_prep_w(
    const float* __restrict__ w, __bf16* __restrict__ Bbf, float* __restrict__ c0)
{
    const int wid  = threadIdx.x >> 5;
    const int lane = threadIdx.x & 31;
    const int row  = blockIdx.x * 8 + wid;          // 4000 blocks * 8 = 32000

    const float* src = w + (size_t)row * W_STRIDE + 1 + lane * 8;
    float s = 0.0f;
    bf16x8 o;
    #pragma unroll
    for (int i = 0; i < 8; ++i) {
        float v = src[i];
        s += v * v;
        o[i] = (__bf16)v;
    }
    *(bf16x8*)(Bbf + (size_t)row * K_DIM + lane * 8) = o;

    #pragma unroll
    for (int off = 16; off > 0; off >>= 1)
        s += __shfl_xor(s, off, 32);
    if (lane == 0)
        c0[row] = __fsqrt_rn(1.0f + s);
}

// ---------------------------------------------------------------------------
// Stage one K-step: A tile 128x32 bf16 + B tile 128x32 bf16 -> LDS (padded).
// 32 chunks of (8 rows x 64B) handled by 8 waves * 4 chunks; 4 lanes per row.
// ---------------------------------------------------------------------------
__device__ __forceinline__ void stage_tiles(
    const __bf16* __restrict__ Ag, const __bf16* __restrict__ Bg, int kk,
    __bf16* lA, __bf16* lB, int wave, int lane)
{
    const int rsub = lane >> 2;        // 0..7  row within chunk
    const int col8 = (lane & 3) * 8;   // 0,8,16,24 (bf16 elems; 16B each)
    #pragma unroll
    for (int i = 0; i < 4; ++i) {
        const int chunk = wave * 4 + i;     // 0..31
        const int isB   = chunk >> 4;       // 0..15 -> A, 16..31 -> B
        const int row   = (chunk & 15) * 8 + rsub;   // 0..127
        const __bf16* g = (isB ? Bg : Ag) + (size_t)row * K_DIM + kk + col8;
        __bf16*       l = (isB ? lB : lA) + row * LDS_STRIDE + col8;
        async_copy_b128(g, l);
    }
}

// ---------------------------------------------------------------------------
// GEMM + fused Lorentz epilogue.
// Block: 256 threads = 8 waves, tile 128(M) x 128(N), LDS double-buffered.
// Wave:  64(M) x 32(N) = 4x2 wmma tiles, K-loop 8 steps of 32.
// Fragment layouts (C/D as in CDNA5 ISA 7.12.2):
//   A 16x32 bf16: lane<16 row M=lane, K 0..15; lane>=16 K 16..31 (contiguous, same as B)
//   B 32x16 bf16: lane<16 col N=lane, K 0..15 in v0..7;   lane>=16 K 16..31 (code's order, not ISA's)
//   C/D f32:      vgpr j, lane -> (M = j + 8*(lane>>4), N = lane&15)
// ---------------------------------------------------------------------------
__global__ __launch_bounds__(256) void lorentz_wmma_gemm(
    const __bf16* __restrict__ A, const __bf16* __restrict__ Bm,
    const float* __restrict__ x0, const float* __restrict__ c0,
    float* __restrict__ out)
{
    __shared__ __align__(16) __bf16 smem[2][2][TILE_ELEMS];   // [buf][A|B]

    const int lane = threadIdx.x & 31;
    const int wave = threadIdx.x >> 5;
    const int wm   = wave & 1;                      // 2 waves along M
    const int wn   = wave >> 1;                     // 4 waves along N
    const int r    = lane & 15;
    const int hi   = lane >> 4;

    const __bf16* Ag = A  + (size_t)(blockIdx.y * 128) * K_DIM;
    const __bf16* Bg = Bm + (size_t)(blockIdx.x * 128) * K_DIM;

    f32x8 acc[4][2];
    #pragma unroll
    for (int mt = 0; mt < 4; ++mt)
        #pragma unroll
        for (int nt = 0; nt < 2; ++nt)
            acc[mt][nt] = (f32x8){0.f,0.f,0.f,0.f,0.f,0.f,0.f,0.f};

    // Prologue: stage K-step 0 into buffer 0.
    stage_tiles(Ag, Bg, 0, &smem[0][0][0], &smem[0][1][0], wave, lane);

    #pragma unroll
    for (int step = 0; step < 8; ++step) {
        // Kick off next K-step DMA into the other buffer.
        if (step < 7) {
            const int nb = (step + 1) & 1;
            stage_tiles(Ag, Bg, (step + 1) * 32,
                        &smem[nb][0][0], &smem[nb][1][0], wave, lane);
            wait_async_le4();   // 4 newest = next stage; current stage retired
        } else {
            wait_async_0();
        }
        __syncthreads();        // current buffer visible to all waves

        const int cb = step & 1;
        const __bf16* lA = &smem[cb][0][0] + (wm * 64) * LDS_STRIDE;
        const __bf16* lB = &smem[cb][1][0] + (wn * 32) * LDS_STRIDE;

        bf16x16 bfrag[2];
        #pragma unroll
        for (int nt = 0; nt < 2; ++nt) {
            const __bf16* bp = lB + (nt * 16 + r) * LDS_STRIDE + hi * 16;
            bf16x8 b0 = *(const bf16x8*)bp;
            bf16x8 b1 = *(const bf16x8*)(bp + 8);
            bfrag[nt] = __builtin_shufflevector(b0, b1,
                0,1,2,3,4,5,6,7,8,9,10,11,12,13,14,15);
        }
        #pragma unroll
        for (int mt = 0; mt < 4; ++mt) {
            const __bf16* ap = lA + (mt * 16 + r) * LDS_STRIDE + hi * 16;
            bf16x8 a0 = *(const bf16x8*)ap;          // K = hi*16     .. +7
            bf16x8 a1 = *(const bf16x8*)(ap + 8);    // K = hi*16 + 8 .. +7
            bf16x16 afrag = __builtin_shufflevector(a0, a1,
                0,1,2,3,4,5,6,7,8,9,10,11,12,13,14,15);
            #pragma unroll
            for (int nt = 0; nt < 2; ++nt) {
                acc[mt][nt] = __builtin_amdgcn_wmma_f32_16x16x32_bf16(
                    /*neg_a=*/false, afrag,
                    /*neg_b=*/false, bfrag[nt],
                    /*c_mod=*/(short)0, acc[mt][nt],
                    /*reuse_a=*/false, /*reuse_b=*/false);
            }
        }
        __syncthreads();        // all reads done before buffer is overwritten
    }

    // Fused epilogue: inner = x0*c0 - dot ; out = -arccosh(max(inner, 1+eps))
    const int mBase = blockIdx.y * 128 + wm * 64;
    const int nBase = blockIdx.x * 128 + wn * 32;
    #pragma unroll
    for (int mt = 0; mt < 4; ++mt) {
        #pragma unroll
        for (int nt = 0; nt < 2; ++nt) {
            const int n   = nBase + nt * 16 + r;
            const float c = c0[n];
            #pragma unroll
            for (int j = 0; j < 8; ++j) {
                const int m = mBase + mt * 16 + hi * 8 + j;
                float inner = x0[m] * c - acc[mt][nt][j];
                float t = fmaxf(inner, EPS_1);
                float d = __logf(t + __fsqrt_rn(t * t - 1.0f));
                __builtin_nontemporal_store(-d, &out[(size_t)m * N_DIM + n]);
            }
        }
    }
}

// ---------------------------------------------------------------------------
// Host-side launcher
// ---------------------------------------------------------------------------
extern "C" void kernel_launch(void* const* d_in, const int* in_sizes, int n_in,
                              void* d_out, int out_size, void* d_ws, size_t ws_size,
                              hipStream_t stream) {
    (void)in_sizes; (void)n_in; (void)out_size; (void)ws_size;

    const float* x = (const float*)d_in[0];   // (2,2048,256) f32
    const float* w = (const float*)d_in[1];   // (32000,257)  f32
    float* out = (float*)d_out;               // (2,2048,32000) f32

    char* ws = (char*)d_ws;
    const size_t A_BYTES = (size_t)M_DIM * K_DIM * 2;   // 2,097,152
    const size_t B_BYTES = (size_t)N_DIM * K_DIM * 2;   // 16,384,000
    __bf16* Abf = (__bf16*)(ws);
    __bf16* Bbf = (__bf16*)(ws + A_BYTES);
    float*  x0  = (float*)(ws + A_BYTES + B_BYTES);
    float*  c0  = (float*)(ws + A_BYTES + B_BYTES + (size_t)M_DIM * 4);

    lorentz_prep_x<<<M_DIM / 8, 256, 0, stream>>>(x, Abf, x0);
    lorentz_prep_w<<<N_DIM / 8, 256, 0, stream>>>(w, Bbf, c0);

    dim3 grid(N_DIM / 128, M_DIM / 128);  // (250, 32)
    lorentz_wmma_gemm<<<grid, 256, 0, stream>>>(Abf, Bbf, x0, c0, out);
}
